// Attention_49478023250074
// MI455X (gfx1250) — compile-verified, matrix operand order checked by reading, not run on hardware
//
#include <hip/hip_runtime.h>
#include <hip/hip_bf16.h>

// Problem constants (match the JAX reference)
constexpr int kB  = 4;
constexpr int kS  = 2048;
constexpr int kD  = 768;
constexpr int kH  = 12;
constexpr int kDh = 64;

typedef __bf16 v16bf __attribute__((ext_vector_type(16)));
typedef float  v8f   __attribute__((ext_vector_type(8)));

typedef unsigned int u32x4 __attribute__((ext_vector_type(4)));
typedef int          i32x4 __attribute__((ext_vector_type(4)));
typedef int          i32x8 __attribute__((ext_vector_type(8)));

// LDS row strides (elements). 16-byte multiples, non-power-of-two dword
// strides so 16-lane ds_load_b128 row fan-outs spread across banks.
constexpr int kLd32 = 40;   // for 32-wide (k) tiles
constexpr int kLd64 = 72;   // for 64-wide tiles (matches TDM pad: 32dw + 4dw)

// ---------------------------------------------------------------------------
// WMMA helper: D = A(16x32 bf16) * B(32x16 bf16) + C(16x16 f32)
// ---------------------------------------------------------------------------
__device__ __forceinline__ v8f wmma_bf16(v16bf a, v16bf b, v8f c) {
  return __builtin_amdgcn_wmma_f32_16x16x32_bf16(
      false, a, false, b, (short)0, c, false, false);
}

// ---------------------------------------------------------------------------
// Fragment loader (CDNA5 ISA §7.12.2, wave32). Both A and B use
// lane = m (or n) in [0,16), k-half = (lane/16)*8; VGPR v<4 -> k = half+2v,
// VGPR v>=4 -> k = 16+half+2v. Source tile must store k contiguously per
// row so each half lowers to one ds_load_b128 / global_load_b128.
// ---------------------------------------------------------------------------
__device__ __forceinline__ v16bf load_frag_kcontig(const __bf16* base, int ld) {
  const int lane = threadIdx.x & 31;
  const int r    = lane & 15;
  const int kh   = (lane >> 4) << 3;
  const __bf16* p = base + r * ld;
  v16bf f;
#pragma unroll
  for (int v = 0; v < 4; ++v) {
    f[2*v]     = p[kh + 2*v];
    f[2*v + 1] = p[kh + 2*v + 1];
    f[2*v + 8] = p[16 + kh + 2*v];
    f[2*v + 9] = p[16 + kh + 2*v + 1];
  }
  return f;
}

__device__ __forceinline__ unsigned pack2f(float lo, float hi) {
  union { __bf16 b[2]; unsigned u; } u;
  u.b[0] = (__bf16)lo; u.b[1] = (__bf16)hi;
  return u.u;
}
__device__ __forceinline__ unsigned pack2b(__bf16 lo, __bf16 hi) {
  union { __bf16 b[2]; unsigned u; } u;
  u.b[0] = lo; u.b[1] = hi;
  return u.u;
}

// ---------------------------------------------------------------------------
// Tensor Data Mover: load a 64(key) x 64(dh) bf16 tile Global->LDS, with the
// TDM pad feature inserting 4 DWORDs every 32 DWORDs so rows land with a
// 72-element stride. D# layout per CDNA5 ISA.
// ---------------------------------------------------------------------------
__device__ __forceinline__ void tdm_load_tile64x64_bf16(unsigned lds_byte_off,
                                                        const void* gptr) {
  const unsigned long long ga = (unsigned long long)(size_t)gptr;
  u32x4 g0;
  g0[0] = 1u;                                   // count=1, user mode
  g0[1] = lds_byte_off;                         // lds_addr
  g0[2] = (unsigned)(ga & 0xffffffffull);       // global_addr[31:0]
  g0[3] = (unsigned)((ga >> 32) & 0x1ffffffull) // global_addr[56:32]
        | (2u << 30);                           // type = 2 ("image")
  i32x8 g1;
  g1[0] = (int)((1u << 16)      // data_size = 1 (2 bytes)
              | (1u << 20)      // pad_enable
              | (4u << 22)      // pad_interval: every 32 DWORDs
              | (3u << 25));    // pad_amount: 4 DWORDs
  g1[1] = (int)(64u << 16);     // tensor_dim0 = 64 (dh, contiguous)
  g1[2] = (int)(2048u << 16);   // tensor_dim1 = 2048 (keys)
  g1[3] = (int)(64u << 16);     // tile_dim0 = 64
  g1[4] = 64;                   // tile_dim1 = 64, tile_dim2 = 0
  g1[5] = 64;                   // tensor_dim0_stride = 64
  g1[6] = 0;
  g1[7] = 0;
  const i32x4 z4 = {0, 0, 0, 0};
#if defined(__clang_major__) && (__clang_major__ >= 23)
  const i32x8 z8 = {0, 0, 0, 0, 0, 0, 0, 0};
  __builtin_amdgcn_tensor_load_to_lds(g0, g1, z4, z4, z8, 0);
#else
  __builtin_amdgcn_tensor_load_to_lds(g0, g1, z4, z4, 0);
#endif
}

__device__ __forceinline__ unsigned lds_off(const void* p) {
  // Generic pointers to LDS carry the byte offset in the low 32 bits
  // (shared aperture occupies addr[63:32]).
  return (unsigned)(unsigned long long)(size_t)p;
}

// ---------------------------------------------------------------------------
// Kernel 1: QKV projection. q/k = (resid+pos) @ W + b ; v = resid @ W + b.
// Q pre-scaled by 1/sqrt(Dh). Output layout [B, H, S, Dh] bf16.
// grid = (128, 36), block = 128 (4 waves), 64x64 tile per block.
// ---------------------------------------------------------------------------
__global__ __launch_bounds__(128) void qkv_proj_kernel(
    const float* __restrict__ resid, const float* __restrict__ pos,
    const float* __restrict__ Wq, const float* __restrict__ Wk,
    const float* __restrict__ Wv,
    const float* __restrict__ bq, const float* __restrict__ bk,
    const float* __restrict__ bv,
    __bf16* __restrict__ Qo, __bf16* __restrict__ Ko, __bf16* __restrict__ Vo)
{
  const int which = blockIdx.y / kH;   // 0=Q, 1=K, 2=V
  const int h     = blockIdx.y % kH;
  const int m0    = blockIdx.x * 64;

  const float*  W    = (which == 0) ? Wq : (which == 1) ? Wk : Wv;
  const float*  bias = (which == 0) ? bq : (which == 1) ? bk : bv;
  __bf16*       Out  = (which == 0) ? Qo : (which == 1) ? Ko : Vo;

  __shared__ __align__(16) __bf16 Xs[64 * kLd32];   // activations [m][k]
  __shared__ __align__(16) __bf16 Wt[64 * kLd32];   // weights transposed [n][k]

  const int tid  = threadIdx.x;
  const int w    = tid >> 5;
  const int lane = tid & 31;

  v8f acc[4] = {};

  for (int kk = 0; kk < kD; kk += 32) {
    // Activations: 64x32, packed bf16 pair stores (u32).
    for (int j = tid; j < 64 * 16; j += 128) {
      const int cp = j & 15, r = j >> 4;
      const size_t g = (size_t)(m0 + r) * kD + kk + 2 * cp;
      float x0 = resid[g], x1 = resid[g + 1];
      if (which < 2) { x0 += pos[g]; x1 += pos[g + 1]; }
      ((unsigned*)Xs)[r * (kLd32 / 2) + cp] = pack2f(x0, x1);
    }
    // Weights transposed: Wt[n][k]; n fastest over threads -> coalesced reads.
    for (int j = tid; j < 64 * 16; j += 128) {
      const int n = j & 63, kp = j >> 6;
      const size_t g = (size_t)h * kD * kDh + (size_t)(kk + 2 * kp) * kDh + n;
      ((unsigned*)Wt)[n * (kLd32 / 2) + kp] = pack2f(W[g], W[g + kDh]);
    }
    __syncthreads();

    const v16bf a = load_frag_kcontig(Xs + (w * 16) * kLd32, kLd32);
#pragma unroll
    for (int sub = 0; sub < 4; ++sub) {
      const v16bf bb = load_frag_kcontig(Wt + (sub * 16) * kLd32, kLd32);
      acc[sub] = wmma_bf16(a, bb, acc[sub]);
    }
    __syncthreads();
  }

  const float qscale = (which == 0) ? 0.125f : 1.0f;   // 1/sqrt(64)
  const int n_lo = lane & 15;
  const int mb   = (lane >> 4) * 8;
#pragma unroll
  for (int sub = 0; sub < 4; ++sub) {
    const int n = sub * 16 + n_lo;
    const float bval = bias[h * kDh + n];
#pragma unroll
    for (int v = 0; v < 8; ++v) {
      const int row = m0 + w * 16 + mb + v;        // row = b*S + s
      const int b   = row >> 11;
      const int s   = row & 2047;
      Out[(((size_t)b * kH + h) * kS + s) * kDh + n] =
          (__bf16)((acc[sub][v] + bval) * qscale);
    }
  }
}

// ---------------------------------------------------------------------------
// Kernel 2: causal flash attention; one (b, h, 64 q-rows) per block.
// K tile fetched by the Tensor Data Mover (padded to 72-elem rows); V tile
// transposed into [dh][key] so the PV B-fragments are k-contiguous.
// ---------------------------------------------------------------------------
__global__ __launch_bounds__(128) void flash_attn_kernel(
    const __bf16* __restrict__ Qb, const __bf16* __restrict__ Kb,
    const __bf16* __restrict__ Vb, __bf16* __restrict__ Zb)
{
  const int bh = blockIdx.y;
  const int b  = bh / kH;
  const int h  = bh % kH;
  const int q0 = blockIdx.x * 64;

  const size_t base = ((size_t)b * kH + h) * (size_t)kS * kDh;
  const __bf16* Qp = Qb + base;
  const __bf16* Kp = Kb + base;
  const __bf16* Vp = Vb + base;

  __shared__ __align__(16) __bf16 Ks[64 * kLd64];      // [key][dh]   (via TDM)
  __shared__ __align__(16) __bf16 Vt[64 * kLd64];      // [dh][key]   (transposed)
  __shared__ __align__(16) __bf16 Ps[4][16 * kLd64];   // per-wave P relayout

  const int tid  = threadIdx.x;
  const int w    = tid >> 5;
  const int lane = tid & 31;
  const int n_lo = lane & 15;
  const int mb   = (lane >> 4) * 8;

  // Q fragments, kept in registers for the whole pass.
  const __bf16* Qtile = Qp + (size_t)(q0 + w * 16) * kDh;
  const v16bf qa0 = load_frag_kcontig(Qtile,      kDh);  // dh 0..31
  const v16bf qa1 = load_frag_kcontig(Qtile + 32, kDh);  // dh 32..63

  v8f o[4] = {};
  float mrun[8], lrun[8];
#pragma unroll
  for (int v = 0; v < 8; ++v) { mrun[v] = -3.0e38f; lrun[v] = 0.0f; }

  const unsigned ks_off = lds_off(Ks);
  const int ktiles = blockIdx.x + 1;   // causal: only key tiles <= q tile

  for (int kt = 0; kt < ktiles; ++kt) {
    // K tile: one TDM descriptor issued by wave 0 (EXEC-independent DMA).
    if (w == 0) {
      tdm_load_tile64x64_bf16(ks_off, Kp + (size_t)kt * 64 * kDh);
    }
    // V tile: cooperative transpose [key][dh] -> Vt[dh][key], packed u32.
    for (int j = tid; j < 64 * 32; j += 128) {
      const int dh = j & 63, kp = j >> 6;                 // keys 2kp, 2kp+1
      const __bf16* vg = Vp + (size_t)(kt * 64 + 2 * kp) * kDh + dh;
      ((unsigned*)Vt)[dh * (kLd64 / 2) + kp] = pack2b(vg[0], vg[kDh]);
    }
    if (kt + 1 < ktiles) {
      __builtin_prefetch(Kp + (size_t)(kt + 1) * 64 * kDh + tid * 32, 0, 1);
      __builtin_prefetch(Vp + (size_t)(kt + 1) * 64 * kDh + tid * 32, 0, 1);
    }
    if (w == 0) __builtin_amdgcn_s_wait_tensorcnt(0);
    __syncthreads();

    // Scores: S = Q * K^T (B frags k-contiguous from [key][dh] rows).
    v8f sc[4];
#pragma unroll
    for (int sub = 0; sub < 4; ++sub) {
      v8f z = {};
      z = wmma_bf16(qa0, load_frag_kcontig(Ks + (sub * 16) * kLd64,      kLd64), z);
      z = wmma_bf16(qa1, load_frag_kcontig(Ks + (sub * 16) * kLd64 + 32, kLd64), z);
      sc[sub] = z;
    }

    // Causal mask: only the diagonal tile has k > q entries.
    if (kt == ktiles - 1) {
#pragma unroll
      for (int sub = 0; sub < 4; ++sub) {
        const int kg = kt * 64 + sub * 16 + n_lo;
#pragma unroll
        for (int v = 0; v < 8; ++v) {
          const int qg = q0 + w * 16 + mb + v;
          if (kg > qg) sc[sub][v] = -3.0e38f;
        }
      }
    }

    // Online softmax; xor-shuffles 8/4/2/1 reduce inside each 16-lane half.
#pragma unroll
    for (int v = 0; v < 8; ++v) {
      float rm = fmaxf(fmaxf(sc[0][v], sc[1][v]), fmaxf(sc[2][v], sc[3][v]));
      rm = fmaxf(rm, __shfl_xor(rm, 8));
      rm = fmaxf(rm, __shfl_xor(rm, 4));
      rm = fmaxf(rm, __shfl_xor(rm, 2));
      rm = fmaxf(rm, __shfl_xor(rm, 1));
      const float mnew  = fmaxf(mrun[v], rm);
      const float alpha = __expf(mrun[v] - mnew);
      float rs = 0.0f;
#pragma unroll
      for (int sub = 0; sub < 4; ++sub) {
        const float p = __expf(sc[sub][v] - mnew);
        sc[sub][v] = p;
        rs += p;
      }
      rs += __shfl_xor(rs, 8);
      rs += __shfl_xor(rs, 4);
      rs += __shfl_xor(rs, 2);
      rs += __shfl_xor(rs, 1);
      lrun[v] = lrun[v] * alpha + rs;
      mrun[v] = mnew;
#pragma unroll
      for (int sub = 0; sub < 4; ++sub) o[sub][v] *= alpha;
    }

    // Relayout P: C-layout f32 -> row-major bf16 (per-wave region) -> A frags.
    __bf16* Pw = Ps[w];
#pragma unroll
    for (int sub = 0; sub < 4; ++sub)
#pragma unroll
      for (int v = 0; v < 8; ++v)
        Pw[(mb + v) * kLd64 + sub * 16 + n_lo] = (__bf16)sc[sub][v];

    const v16bf pa0 = load_frag_kcontig(Pw,      kLd64);  // keys 0..31
    const v16bf pa1 = load_frag_kcontig(Pw + 32, kLd64);  // keys 32..63

    // O += P * V  (Vt[dh][key]: k-contiguous B frags).
#pragma unroll
    for (int sub = 0; sub < 4; ++sub) {
      o[sub] = wmma_bf16(pa0, load_frag_kcontig(Vt + (sub * 16) * kLd64,      kLd64), o[sub]);
      o[sub] = wmma_bf16(pa1, load_frag_kcontig(Vt + (sub * 16) * kLd64 + 32, kLd64), o[sub]);
    }
    __syncthreads();
  }

  // Normalize and emit Z as bf16 [B, S, H*Dh].
#pragma unroll
  for (int v = 0; v < 8; ++v) {
    const float inv = 1.0f / lrun[v];
    const int s = q0 + w * 16 + mb + v;
#pragma unroll
    for (int sub = 0; sub < 4; ++sub) {
      Zb[((size_t)b * kS + s) * kD + h * kDh + sub * 16 + n_lo] =
          (__bf16)(o[sub][v] * inv);
    }
  }
}

// ---------------------------------------------------------------------------
// Kernel 3: output projection. out = Z[8192,768] @ W_O(flat [768,768]) + b_O.
// ---------------------------------------------------------------------------
__global__ __launch_bounds__(128) void out_proj_kernel(
    const __bf16* __restrict__ Zb, const float* __restrict__ Wo,
    const float* __restrict__ bo, float* __restrict__ Out)
{
  const int m0 = blockIdx.x * 64;
  const int n0 = blockIdx.y * 64;

  __shared__ __align__(16) __bf16 Zs[64 * kLd32];
  __shared__ __align__(16) __bf16 Wt[64 * kLd32];

  const int tid  = threadIdx.x;
  const int w    = tid >> 5;
  const int lane = tid & 31;

  v8f acc[4] = {};

  for (int kk = 0; kk < kD; kk += 32) {
    // Z tile (already bf16): packed u32 copies.
    for (int j = tid; j < 64 * 16; j += 128) {
      const int cp = j & 15, r = j >> 4;
      ((unsigned*)Zs)[r * (kLd32 / 2) + cp] =
          ((const unsigned*)(Zb + (size_t)m0 * kD + kk))[(size_t)r * (kD / 2) + cp];
    }
    // W_O tile transposed: Wt[n][k].
    for (int j = tid; j < 64 * 16; j += 128) {
      const int n = j & 63, kp = j >> 6;
      const size_t g = (size_t)(kk + 2 * kp) * kD + n0 + n;
      ((unsigned*)Wt)[n * (kLd32 / 2) + kp] = pack2f(Wo[g], Wo[g + kD]);
    }
    __syncthreads();

    const v16bf a = load_frag_kcontig(Zs + (w * 16) * kLd32, kLd32);
#pragma unroll
    for (int sub = 0; sub < 4; ++sub)
      acc[sub] = wmma_bf16(a, load_frag_kcontig(Wt + (sub * 16) * kLd32, kLd32),
                           acc[sub]);
    __syncthreads();
  }

  const int n_lo = lane & 15;
  const int mb   = (lane >> 4) * 8;
#pragma unroll
  for (int sub = 0; sub < 4; ++sub) {
    const int n = n0 + sub * 16 + n_lo;
    const float bval = bo[n];
#pragma unroll
    for (int v = 0; v < 8; ++v) {
      const int row = m0 + w * 16 + mb + v;
      Out[(size_t)row * kD + n] = acc[sub][v] + bval;
    }
  }
}

// ---------------------------------------------------------------------------
// Launch. Inputs (all f32, setup_inputs order): resid, pos, W_Q, W_K, W_V,
// W_O, b_Q, b_K, b_V, b_O. Output f32 [B,S,D]. Workspace: bf16 Q|K|V
// ([B,H,S,Dh] each) then Z ([B,S,H*Dh]); total ~48 MiB.
// ---------------------------------------------------------------------------
extern "C" void kernel_launch(void* const* d_in, const int* in_sizes, int n_in,
                              void* d_out, int out_size, void* d_ws, size_t ws_size,
                              hipStream_t stream) {
  (void)in_sizes; (void)n_in; (void)out_size; (void)ws_size;

  const float* resid = (const float*)d_in[0];
  const float* pos   = (const float*)d_in[1];
  const float* Wq    = (const float*)d_in[2];
  const float* Wk    = (const float*)d_in[3];
  const float* Wv    = (const float*)d_in[4];
  const float* Wo    = (const float*)d_in[5];
  const float* bq    = (const float*)d_in[6];
  const float* bk    = (const float*)d_in[7];
  const float* bv    = (const float*)d_in[8];
  const float* bo    = (const float*)d_in[9];
  float* out = (float*)d_out;

  const size_t per = (size_t)kB * kH * kS * kDh;  // 6,291,456 elements
  __bf16* Qb = (__bf16*)d_ws;
  __bf16* Kb = Qb + per;
  __bf16* Vb = Kb + per;
  __bf16* Zb = Vb + per;

  qkv_proj_kernel<<<dim3((kB * kS) / 64, 3 * kH), 128, 0, stream>>>(
      resid, pos, Wq, Wk, Wv, bq, bk, bv, Qb, Kb, Vb);

  flash_attn_kernel<<<dim3(kS / 64, kB * kH), 128, 0, stream>>>(Qb, Kb, Vb, Zb);

  out_proj_kernel<<<dim3((kB * kS) / 64, kD / 64), 128, 0, stream>>>(
      Zb, Wo, bo, out);
}
